// RouterOursNewToken_27788438405469
// MI455X (gfx1250) — matrix operand order repaired and checked by reading, not compiled, not run on hardware
//
#include <hip/hip_runtime.h>
#include <hip/hip_bf16.h>
#include <math.h>

typedef __bf16 v16bf __attribute__((ext_vector_type(16)));
typedef float  v8f   __attribute__((ext_vector_type(8)));

#define B_  32
#define H_  12
#define L_  512
#define D_  768
#define KTOP 255
#define HD  64
#define NSLICE 16          // row-slices for the scores reduction

// ---------------- workspace layout (floats), two overlapping phases ----------------
// Phase 1 (K1..K4):  imp_part [0, 262144)   idx [262144, 270336)
// Phase 2 (K5a..K8): sent [0, 24576) u [24576,417792) qbk [417792,418304)
//                    logits [418304,614912) hbar [614912,909824)
// Safe because the stream serializes: imp_part dead after K3, idx dead after K4.
#define OFF_PART   0
#define OFF_IDX    262144
#define OFF_SENT   0
#define OFF_U      24576
#define OFF_QBK    417792
#define OFF_LOGITS 418304
#define OFF_HBAR   614912

// ---------------- K1: partial[b,s,l] = sum_{rows in slice s} scores[b,row,l]*amf[b,i] ----
// grid (B, NSLICE), block 256.  float4 loads: 256 threads read 2 full rows (4KB) / iter.
// Deterministic: no atomics; per-slice partials summed in fixed order by K3.
__global__ void k1_scores_reduce(const float* __restrict__ scores,
                                 const float* __restrict__ am,
                                 float* __restrict__ imp_part) {
    __shared__ float  samf[L_];
    __shared__ float4 sacc[256];
    int b = blockIdx.x, s = blockIdx.y, t = threadIdx.x;
    samf[t]       = (am[b * L_ + t]       > -10.0f) ? 1.0f : 0.0f;
    samf[t + 256] = (am[b * L_ + t + 256] > -10.0f) ? 1.0f : 0.0f;
    __syncthreads();
    int rowbase = s * (H_ * L_ / NSLICE);          // 384 rows per slice (flattened h*512+i)
    int rpar = t >> 7;                             // which of the 2 rows this iteration
    int c4 = (t & 127) * 4;                        // l-column base
    const float4* p = (const float4*)(scores +
        ((size_t)b * (H_ * L_) + rowbase + rpar) * L_ + c4);
    float4 acc = {0.f, 0.f, 0.f, 0.f};
    for (int i = 0; i < 192; ++i) {                // 192 iters x 2 rows = 384 rows
        __builtin_prefetch(p + (size_t)(i + 8) * 256, 0, 1);   // global_prefetch_b8
        float4 v = p[(size_t)i * 256];
        float w = samf[(rowbase + 2 * i + rpar) & (L_ - 1)];
        acc.x += v.x * w; acc.y += v.y * w; acc.z += v.z * w; acc.w += v.w * w;
    }
    sacc[t] = acc;
    __syncthreads();
    if (t < 128) {
        float4 a = sacc[t], o = sacc[t + 128];
        a.x += o.x; a.y += o.y; a.z += o.z; a.w += o.w;
        *(float4*)(imp_part + ((size_t)(b * NSLICE + s)) * L_ + t * 4) = a;
    }
}

// ---------------- K3: exact top-255 (stable), mask, tome ----------------
// one block of 512 threads per batch; sums the 16 slice-partials in fixed order
__global__ void k3_topk(const float* __restrict__ imp_part,
                        const float* __restrict__ am,
                        int* __restrict__ idx_out,
                        float* __restrict__ out_mask,
                        float* __restrict__ out_tome) {
    __shared__ float sv[L_];
    __shared__ int   sc[L_];
    int b = blockIdx.x, t = threadIdx.x;
    float amv = am[b * L_ + t];
    float amf = (amv > -10.0f) ? 1.0f : 0.0f;
    float v = 0.0f;
    #pragma unroll
    for (int s = 0; s < NSLICE; ++s)
        v += imp_part[((size_t)(b * NSLICE + s)) * L_ + t];
    v *= amf * (1.0f / (H_ * L_));
    if (t == 0) v = __builtin_inff();      // always keep CLS
    sv[t] = v;
    __syncthreads();
    int rank = 0;
    for (int j = 0; j < L_; ++j) {
        float vj = sv[j];
        rank += (vj > v) || (vj == v && j < t);   // total order -> exactly KTOP selected
    }
    int sel = (rank < KTOP) ? 1 : 0;
    sc[t] = sel;
    __syncthreads();
    // inclusive Hillis-Steele scan over 512
    for (int off = 1; off < L_; off <<= 1) {
        int add = (t >= off) ? sc[t - off] : 0;
        __syncthreads();
        sc[t] += add;
        __syncthreads();
    }
    if (sel) {
        int pos = sc[t] - 1;               // stable by index => ascending order
        idx_out[b * KTOP + pos]  = t;
        out_mask[b * 256 + pos]  = amv;
    }
    if (t == 0)  out_mask[b * 256 + KTOP] = 0.0f;
    if (t < 256) out_tome[b * 256 + t] = 1.0f;
}

// ---------------- K4: gather preserved rows ----------------
// grid B*255 blocks, 192 threads (float4 copy of 768 floats)
__global__ void k4_gather(const float* __restrict__ hidden,
                          const int* __restrict__ idx,
                          float* __restrict__ out_tok) {
    int blk = blockIdx.x;
    int b = blk / KTOP, k = blk % KTOP;
    int src = idx[b * KTOP + k];
    const float4* s = (const float4*)(hidden + ((size_t)(b * L_ + src)) * D_);
    float4* d = (float4*)(out_tok + ((size_t)(b * 256 + k)) * D_);
    d[threadIdx.x] = s[threadIdx.x];
}

// ---------------- K5a: sentences = softmax(am) @ hidden ----------------
// grid B*3 blocks, 256 threads (each block does 256 feature dims)
__global__ void k5a_sentence(const float* __restrict__ hidden,
                             const float* __restrict__ am,
                             float* __restrict__ sent) {
    __shared__ float s_att[L_];
    __shared__ float sred[256];
    int b = blockIdx.x / 3, dc = blockIdx.x % 3, t = threadIdx.x;
    float a0 = am[b * L_ + t], a1 = am[b * L_ + 256 + t];
    sred[t] = fmaxf(a0, a1);
    __syncthreads();
    for (int off = 128; off > 0; off >>= 1) {
        if (t < off) sred[t] = fmaxf(sred[t], sred[t + off]);
        __syncthreads();
    }
    float mx = sred[0];
    __syncthreads();
    float e0 = expf(a0 - mx), e1 = expf(a1 - mx);
    sred[t] = e0 + e1;
    __syncthreads();
    for (int off = 128; off > 0; off >>= 1) {
        if (t < off) sred[t] += sred[t + off];
        __syncthreads();
    }
    float inv = 1.0f / sred[0];
    s_att[t] = e0 * inv;
    s_att[t + 256] = e1 * inv;
    __syncthreads();
    int d = dc * 256 + t;
    float acc = 0.0f;
    for (int l = 0; l < L_; ++l)
        acc += s_att[l] * hidden[((size_t)(b * L_ + l)) * D_ + d];
    sent[b * D_ + d] = acc;
}

// ---------------- K5b: q = sent@Wq+bq ; u[n,:] = Wk[:,n-slice]@q[n] ; qbk ----------
// one block of 256 per batch
__global__ void k5b_q_u(const float* __restrict__ sent,
                        const float* __restrict__ Wq, const float* __restrict__ bq,
                        const float* __restrict__ Wk, const float* __restrict__ bk,
                        float* __restrict__ u, float* __restrict__ qbk) {
    __shared__ float s_sent[D_];
    __shared__ float s_q[D_];
    int b = blockIdx.x, t = threadIdx.x;
    for (int i = t; i < D_; i += 256) s_sent[i] = sent[b * D_ + i];
    __syncthreads();
    for (int rep = 0; rep < 3; ++rep) {
        int j = rep * 256 + t;
        float acc = bq[j];
        for (int d = 0; d < D_; ++d) acc += s_sent[d] * Wq[(size_t)d * D_ + j];
        s_q[j] = acc;
    }
    __syncthreads();
    for (int rep = 0; rep < 3; ++rep) {
        int din = rep * 256 + t;
        for (int n = 0; n < 16; ++n) {
            float acc = 0.0f;
            if (n < H_) {
                const float* wr = Wk + (size_t)din * D_ + n * HD;
                const float* qq = s_q + n * HD;
                #pragma unroll 8
                for (int j = 0; j < HD; ++j) acc += wr[j] * qq[j];
            }
            u[((size_t)(b * 16 + n)) * D_ + din] = acc;   // layout [B][16][768]
        }
    }
    if (t < 16) {
        float acc = 0.0f;
        if (t < H_)
            for (int j = 0; j < HD; ++j) acc += bk[t * HD + j] * s_q[t * HD + j];
        qbk[b * 16 + t] = acc;
    }
}

// ---------------- K6: logits[b,h,l] via WMMA bf16 ----------------
// A = hidden rows (M=16 key positions x K=32), B = u (K=32 x N=16 heads, 12 used)
// grid 128 blocks x 256 threads (8 waves); wave -> one 16-row tile, 24 K-steps.
__global__ void k6_logits_wmma(const float* __restrict__ hidden,
                               const float* __restrict__ u,
                               const float* __restrict__ qbk,
                               const float* __restrict__ am,
                               float* __restrict__ logits) {
    int lane = threadIdx.x & 31;
    int wave = threadIdx.x >> 5;
    int b = blockIdx.x >> 2;                       // 4 blocks per batch
    int mtile = (blockIdx.x & 3) * 8 + wave;       // 32 tiles of 16 rows
    int half = lane >> 4;
    int mrow = mtile * 16 + (lane & 15);           // key position for A rows
    int n = lane & 15;                             // head column
    const float* arow = hidden + ((size_t)(b * L_ + mrow)) * D_;
    const float* bbase = u + ((size_t)(b * 16 + n)) * D_ + half * 8;
    v8f acc = {};
    for (int kk = 0; kk < D_ / 32; ++kk) {
        int kbase = kk * 32;
        v16bf a, bf;
        // A: 16-bit A-matrix 16x32 layout (ISA 7.12.2)
        #pragma unroll
        for (int v = 0; v < 8; ++v) {
            int K0 = ((v < 4) ? 0 : 16) + half * 8 + ((v & 3) << 1);
            float2 f = *(const float2*)(arow + kbase + K0);
            a[2 * v]     = (__bf16)f.x;
            a[2 * v + 1] = (__bf16)f.y;
        }
        // B: lane = column, K = (e<8 ? 0 : 16) + half*8 + (e&7) (matches A)
        #pragma unroll
        for (int v = 0; v < 4; ++v) {
            float4 f = *(const float4*)(bbase + kbase + ((v < 2) ? 0 : 16) + (v & 1) * 4);
            bf[4 * v]     = (__bf16)f.x;
            bf[4 * v + 1] = (__bf16)f.y;
            bf[4 * v + 2] = (__bf16)f.z;
            bf[4 * v + 3] = (__bf16)f.w;
        }
        acc = __builtin_amdgcn_wmma_f32_16x16x32_bf16(
            false, a, false, bf, (short)0, acc, false, false);
    }
    float qc = qbk[b * 16 + n];
    #pragma unroll
    for (int r = 0; r < 8; ++r) {
        int l = mtile * 16 + r + half * 8;         // D row = r + half*8
        float lv = (acc[r] + qc) * 0.125f;         // / sqrt(64)
        if (am[b * L_ + l] < -10.0f) lv = -__builtin_inff();
        if (n < H_) logits[((size_t)(b * H_ + n)) * L_ + l] = lv;
    }
}

// ---------------- K7: per-head softmax over l, then hbar = w @ hidden ----------
// grid B*3 blocks, 256 threads
__global__ void k7_softmax_hbar(const float* __restrict__ hidden,
                                const float* __restrict__ logits,
                                float* __restrict__ hbar) {
    __shared__ float s_w[H_ * L_];
    __shared__ float sred[256];
    int b = blockIdx.x / 3, dc = blockIdx.x % 3, t = threadIdx.x;
    for (int i = t; i < H_ * L_; i += 256) s_w[i] = logits[(size_t)b * (H_ * L_) + i];
    __syncthreads();
    for (int h = 0; h < H_; ++h) {
        float x0 = s_w[h * L_ + t], x1 = s_w[h * L_ + 256 + t];
        sred[t] = fmaxf(x0, x1);
        __syncthreads();
        for (int off = 128; off > 0; off >>= 1) {
            if (t < off) sred[t] = fmaxf(sred[t], sred[t + off]);
            __syncthreads();
        }
        float mx = sred[0];
        __syncthreads();
        float e0 = expf(x0 - mx), e1 = expf(x1 - mx);
        sred[t] = e0 + e1;
        __syncthreads();
        for (int off = 128; off > 0; off >>= 1) {
            if (t < off) sred[t] += sred[t + off];
            __syncthreads();
        }
        float inv = 1.0f / sred[0];
        s_w[h * L_ + t] = e0 * inv;
        s_w[h * L_ + 256 + t] = e1 * inv;
        __syncthreads();
    }
    int d = dc * 256 + t;
    float acc[H_];
    #pragma unroll
    for (int h = 0; h < H_; ++h) acc[h] = 0.0f;
    for (int l = 0; l < L_; ++l) {
        float x = hidden[((size_t)(b * L_ + l)) * D_ + d];
        #pragma unroll
        for (int h = 0; h < H_; ++h) acc[h] += s_w[h * L_ + l] * x;
    }
    #pragma unroll
    for (int h = 0; h < H_; ++h)
        hbar[((size_t)(b * H_ + h)) * D_ + d] = acc[h];
}

// ---------------- K8: ctx = hbar@Wv + bv ; new_token = ctx@Wo + bo ----------------
// one block of 256 per batch; writes output row 255
__global__ void k8_newtoken(const float* __restrict__ hbar,
                            const float* __restrict__ Wv, const float* __restrict__ bv,
                            const float* __restrict__ Wo, const float* __restrict__ bo,
                            float* __restrict__ out_tok) {
    __shared__ float s_hbar[H_ * D_];
    __shared__ float s_ctx[D_];
    int b = blockIdx.x, t = threadIdx.x;
    for (int i = t; i < H_ * D_; i += 256) s_hbar[i] = hbar[(size_t)b * (H_ * D_) + i];
    __syncthreads();
    for (int rep = 0; rep < 3; ++rep) {
        int uu = rep * 256 + t;
        int n = uu >> 6;
        float acc = bv[uu];
        const float* hb = s_hbar + n * D_;
        for (int d = 0; d < D_; ++d) acc += hb[d] * Wv[(size_t)d * D_ + uu];
        s_ctx[uu] = acc;
    }
    __syncthreads();
    for (int rep = 0; rep < 3; ++rep) {
        int dout = rep * 256 + t;
        float acc = bo[dout];
        for (int u2 = 0; u2 < D_; ++u2) acc += s_ctx[u2] * Wo[(size_t)u2 * D_ + dout];
        out_tok[((size_t)(b * 256 + KTOP)) * D_ + dout] = acc;
    }
}

// ---------------- launcher ----------------
extern "C" void kernel_launch(void* const* d_in, const int* in_sizes, int n_in,
                              void* d_out, int out_size, void* d_ws, size_t ws_size,
                              hipStream_t stream) {
    const float* hidden = (const float*)d_in[0];
    const float* am     = (const float*)d_in[1];
    const float* scores = (const float*)d_in[2];
    const float* Wq = (const float*)d_in[3];
    const float* bq = (const float*)d_in[4];
    const float* Wk = (const float*)d_in[5];
    const float* bk = (const float*)d_in[6];
    const float* Wv = (const float*)d_in[7];
    const float* bv = (const float*)d_in[8];
    const float* Wo = (const float*)d_in[9];
    const float* bo = (const float*)d_in[10];

    float* ws = (float*)d_ws;
    float* imp_part = ws + OFF_PART;
    int*   idx      = (int*)(ws + OFF_IDX);
    float* sent     = ws + OFF_SENT;
    float* u        = ws + OFF_U;
    float* qbk      = ws + OFF_QBK;
    float* logits   = ws + OFF_LOGITS;
    float* hbar     = ws + OFF_HBAR;

    float* out_tok  = (float*)d_out;                       // [32,256,768]
    float* out_mask = out_tok + (size_t)B_ * 256 * D_;     // [32,256]
    float* out_tome = out_mask + (size_t)B_ * 256;         // [32,256]

    k1_scores_reduce<<<dim3(B_, NSLICE), 256, 0, stream>>>(scores, am, imp_part);
    k3_topk<<<B_, 512, 0, stream>>>(imp_part, am, idx, out_mask, out_tome);
    k4_gather<<<B_ * KTOP, 192, 0, stream>>>(hidden, idx, out_tok);
    k5a_sentence<<<B_ * 3, 256, 0, stream>>>(hidden, am, sent);
    k5b_q_u<<<B_, 256, 0, stream>>>(sent, Wq, bq, Wk, bk, u, qbk);
    k6_logits_wmma<<<128, 256, 0, stream>>>(hidden, u, qbk, am, logits);
    k7_softmax_hbar<<<B_ * 3, 256, 0, stream>>>(hidden, logits, hbar);
    k8_newtoken<<<B_, 256, 0, stream>>>(hbar, Wv, bv, Wo, bo, out_tok);
}
